// BertWordPair_62483184222844
// MI455X (gfx1250) — compile-verified, matrix operand order checked by reading, not run on hardware
//
#include <hip/hip_runtime.h>
#include <hip/hip_bf16.h>
#include <math.h>

typedef __attribute__((ext_vector_type(16))) __bf16 v16bf;
typedef __attribute__((ext_vector_type(8)))  __bf16 v8bf;
typedef __attribute__((ext_vector_type(4)))  __bf16 v4bf;
typedef __attribute__((ext_vector_type(8)))  float  v8f;

#define B_     4
#define T_     8
#define L_     512
#define H_     768
#define C_     6
#define BT_    32
#define M1     16384   // BT_ * L_
#define N1     768     // C_ * 128 (64 q + 64 k per class)
#define DQK    64
#define KSTEPS 24      // H_ / 32

// ---------------------------------------------------------------------------
// Kernel 1: projection GEMM (bf16 WMMA, f32 accum) producing only the q/k
// slices, written as Q/K bf16 tensors laid out [bt][c][l][64].
// Block: 256 threads = 8 waves; tile 128(M) x 64(N); K stepped by 32.
// Double-buffered LDS: one barrier per K-step, global loads of tile k+1
// covered by the WMMAs of tile k.
// ---------------------------------------------------------------------------
#define LDA 40   // padded halves/row (80B: 16B-aligned, bank-spread)
#define LDB 40
#define ASZ (128 * LDA)
#define BSZ (64  * LDB)

__global__ __launch_bounds__(256) void proj_qk_kernel(
    const float* __restrict__ x, const float* __restrict__ W,
    const float* __restrict__ bias,
    __bf16* __restrict__ Qb, __bf16* __restrict__ Kb)
{
  __shared__ __bf16 As[2 * ASZ];
  __shared__ __bf16 Bs[2 * BSZ];

  const int tid  = threadIdx.x;
  const int wave = tid >> 5;
  const int lane = tid & 31;
  const int mt = blockIdx.x / (N1 / 64);
  const int nt = blockIdx.x % (N1 / 64);
  const int m0 = mt * 128;
  const int n0 = nt * 64;

  // Column remap is affine inside a 64-wide tile (64 | 128):
  // n0 = c*128 + r0 with r0 in {0, 64}; real W column = wbase + n.
  const int ccls  = n0 >> 7;
  const int r0    = n0 & 127;           // 0 (q half) or 64 (k half)
  const int wbase = ccls * 256 + r0 + ((r0 >> 6) << 6);

  const int halfrow = lane & 15;        // row (A) / col (B) within 16
  const int kb      = (lane >> 4) << 3; // lanes 16..31 take the +8 k-phase

  // Per-thread staging coordinates (constant over K loop).
  int xr[4], xc[4];                     // x: 4 float4 per thread (128x32)
  #pragma unroll
  for (int q = 0; q < 4; ++q) {
    int e = q * 256 + tid;
    xr[q] = e >> 3;
    xc[q] = (e & 7) << 2;
  }
  int wk[2], wn[2];                     // W: 2 float4 per thread (32x64)
  #pragma unroll
  for (int q = 0; q < 2; ++q) {
    int e = q * 256 + tid;
    wk[q] = e >> 4;
    wn[q] = (e & 15) << 2;
  }

  v8f acc[4];
  #pragma unroll
  for (int jt = 0; jt < 4; ++jt) acc[jt] = (v8f)0.0f;

  float4 xa[4];
  float4 wa[2];

#define LOAD_REGS(K0)                                                          \
  do {                                                                         \
    _Pragma("unroll")                                                          \
    for (int q = 0; q < 4; ++q)                                                \
      xa[q] = *(const float4*)&x[(size_t)(m0 + xr[q]) * H_ + (K0) + xc[q]];    \
    _Pragma("unroll")                                                          \
    for (int q = 0; q < 2; ++q)                                                \
      wa[q] = *(const float4*)&W[(size_t)((K0) + wk[q]) * 1536 + wbase + wn[q]];\
  } while (0)

#define STORE_TILE(P)                                                          \
  do {                                                                         \
    __bf16* Ad = &As[(P) * ASZ];                                               \
    __bf16* Bd = &Bs[(P) * BSZ];                                               \
    _Pragma("unroll")                                                          \
    for (int q = 0; q < 4; ++q) {                                              \
      v4bf t;                                                                  \
      t[0] = (__bf16)xa[q].x; t[1] = (__bf16)xa[q].y;                          \
      t[2] = (__bf16)xa[q].z; t[3] = (__bf16)xa[q].w;                          \
      *(v4bf*)&Ad[xr[q] * LDA + xc[q]] = t;                                    \
    }                                                                          \
    _Pragma("unroll")                                                          \
    for (int q = 0; q < 2; ++q) {                                              \
      Bd[(wn[q] + 0) * LDB + wk[q]] = (__bf16)wa[q].x;                         \
      Bd[(wn[q] + 1) * LDB + wk[q]] = (__bf16)wa[q].y;                         \
      Bd[(wn[q] + 2) * LDB + wk[q]] = (__bf16)wa[q].z;                         \
      Bd[(wn[q] + 3) * LDB + wk[q]] = (__bf16)wa[q].w;                         \
    }                                                                          \
  } while (0)

  // Prologue: tile 0 -> regs -> LDS buf 0; issue tile 1 loads; sync.
  LOAD_REGS(0);
  STORE_TILE(0);
  LOAD_REGS(32);
  __syncthreads();

  for (int it = 0; it < KSTEPS; ++it) {
    const int p = it & 1;
    const __bf16* Ab = &As[p * ASZ];
    const __bf16* Bb = &Bs[p * BSZ];

    // A fragment: two 16B LDS reads (halves 0..7 at k=kb, 8..15 at k=16+kb).
    v16bf a;
    {
      const __bf16* ap = &Ab[(wave * 16 + halfrow) * LDA];
      v8bf lo = *(const v8bf*)(ap + kb);
      v8bf hi = *(const v8bf*)(ap + 16 + kb);
      #pragma unroll
      for (int j = 0; j < 8; ++j) { a[j] = lo[j]; a[8 + j] = hi[j]; }
    }
    #pragma unroll
    for (int jt = 0; jt < 4; ++jt) {
      v16bf b;
      const __bf16* bp = &Bb[(jt * 16 + halfrow) * LDB];
      v8bf lo = *(const v8bf*)(bp + kb);
      v8bf hi = *(const v8bf*)(bp + 16 + kb);
      #pragma unroll
      for (int j = 0; j < 8; ++j) { b[j] = lo[j]; b[8 + j] = hi[j]; }
      acc[jt] = __builtin_amdgcn_wmma_f32_16x16x32_bf16(
          false, a, false, b, (short)0, acc[jt], false, false);
    }

    // After the WMMAs: drain tile it+1 regs into the other LDS buffer and
    // issue tile it+2's loads (they complete under the next iteration).
    if (it + 1 < KSTEPS) {
      STORE_TILE(p ^ 1);
      if (it + 2 < KSTEPS) LOAD_REGS((it + 2) * 32);
    }
    __syncthreads();
  }

  // Epilogue: C/D layout row M = v + 8*(lane>=16), col N = lane&15.
  // Q-vs-K choice is uniform per jt -> pointer select, no branches.
  const int mbase = m0 + wave * 16 + ((lane >> 4) << 3);
  #pragma unroll
  for (int jt = 0; jt < 4; ++jt) {
    int gn   = n0 + jt * 16 + halfrow;
    int r    = gn & 127;
    int roff = r & 63;
    float bv = bias[wbase + (gn & 63)];
    __bf16* dst = (r < 64) ? Qb : Kb;
    #pragma unroll
    for (int v = 0; v < 8; ++v) {
      int m  = mbase + v;
      int bt = m >> 9;
      int l  = m & 511;
      size_t off = ((size_t)(bt * C_ + ccls) * L_ + l) * DQK + roff;
      dst[off] = (__bf16)(acc[jt][v] + bv);
    }
  }
#undef LOAD_REGS
#undef STORE_TILE
}

// ---------------------------------------------------------------------------
// Kernel 2: fused logits (Q K^T via bf16 WMMA, K=64 -> 2 wmma per tile) +
// log-softmax over C=6 + masked weighted NLL + per-(b,t) atomic accumulation.
// Block: 256 threads = 8 waves; tile 64x64; wave owns 2 subtiles x 6 classes.
// ---------------------------------------------------------------------------
__global__ __launch_bounds__(256) void logits_loss_kernel(
    const __bf16* __restrict__ Qb, const __bf16* __restrict__ Kb,
    const int* __restrict__ labels, const int* __restrict__ mask,
    const float* __restrict__ cw,
    float* __restrict__ pA, float* __restrict__ pB)
{
  const int tid  = threadIdx.x;
  const int wave = tid >> 5;
  const int lane = tid & 31;

  const int bt  = blockIdx.x >> 6;
  const int rem = blockIdx.x & 63;
  const int m0  = (rem >> 3) * 64;
  const int n0  = (rem & 7)  * 64;

  const int halfrow = lane & 15;
  const int kb      = (lane >> 4) << 3;
  const int isub0   = wave >> 2;        // subtile rows isub0 and isub0+2
  const int jsub    = wave & 3;         // shared column subtile

  v8f acc[2][C_];
  #pragma unroll
  for (int t = 0; t < 2; ++t)
    #pragma unroll
    for (int c = 0; c < C_; ++c) acc[t][c] = (v8f)0.0f;

  #pragma unroll
  for (int c = 0; c < C_; ++c) {
    const __bf16* qp = Qb + (size_t)(bt * C_ + c) * L_ * DQK;
    const __bf16* kp = Kb + (size_t)(bt * C_ + c) * L_ * DQK;

    // B fragments (shared by both subtiles): row n of K, contiguous in k.
    v16bf bfrag[2];
    {
      const __bf16* bp = kp + (size_t)(n0 + jsub * 16 + halfrow) * DQK + kb;
      #pragma unroll
      for (int kc = 0; kc < 2; ++kc) {
        v8bf lo = *(const v8bf*)(bp + kc * 32);
        v8bf hi = *(const v8bf*)(bp + kc * 32 + 16);
        #pragma unroll
        for (int j = 0; j < 8; ++j) { bfrag[kc][j] = lo[j]; bfrag[kc][8 + j] = hi[j]; }
      }
    }
    #pragma unroll
    for (int t = 0; t < 2; ++t) {
      const int it = isub0 + t * 2;
      const __bf16* ap = qp + (size_t)(m0 + it * 16 + halfrow) * DQK + kb;
      #pragma unroll
      for (int kc = 0; kc < 2; ++kc) {
        v16bf a;
        v8bf lo = *(const v8bf*)(ap + kc * 32);
        v8bf hi = *(const v8bf*)(ap + kc * 32 + 16);
        #pragma unroll
        for (int j = 0; j < 8; ++j) { a[j] = lo[j]; a[8 + j] = hi[j]; }
        acc[t][c] = __builtin_amdgcn_wmma_f32_16x16x32_bf16(
            false, a, false, bfrag[kc], (short)0, acc[t][c], false, false);
      }
    }
  }

  // Epilogue: per element log-sum-exp over 6 classes, NLL, masked weight.
  float s1 = 0.0f, s2 = 0.0f;
  const int n = n0 + jsub * 16 + halfrow;
  #pragma unroll
  for (int t = 0; t < 2; ++t) {
    const int it = isub0 + t * 2;
    const int mbase = m0 + it * 16 + ((lane >> 4) << 3);
    #pragma unroll
    for (int v = 0; v < 8; ++v) {
      int m = mbase + v;
      float lg[C_];
      float mx = acc[t][0][v];
      #pragma unroll
      for (int c = 0; c < C_; ++c) { lg[c] = acc[t][c][v]; mx = fmaxf(mx, lg[c]); }
      float se = 0.0f;
      #pragma unroll
      for (int c = 0; c < C_; ++c) se += __expf(lg[c] - mx);
      float lse = mx + __logf(se);
      size_t idx = ((size_t)bt * L_ + m) * L_ + n;
      int lab = labels[idx];
      float wl = cw[lab] * (float)mask[idx];
      s1 += (lse - lg[lab]) * wl;
      s2 += wl;
    }
  }

  __shared__ float r1[256];
  __shared__ float r2[256];
  r1[tid] = s1; r2[tid] = s2;
  __syncthreads();
  #pragma unroll
  for (int off = 128; off > 0; off >>= 1) {
    if (tid < off) { r1[tid] += r1[tid + off]; r2[tid] += r2[tid + off]; }
    __syncthreads();
  }
  if (tid == 0) {
    atomicAdd(&pA[bt], r1[0]);
    atomicAdd(&pB[bt], r2[0]);
  }
}

// ---------------------------------------------------------------------------
// Kernel 3: per-(b,t) normalize and scalar sum.
// ---------------------------------------------------------------------------
__global__ void final_reduce_kernel(const float* __restrict__ pA,
                                    const float* __restrict__ pB,
                                    float* __restrict__ out)
{
  __shared__ float s[BT_];
  int t = threadIdx.x;
  if (t < BT_) s[t] = pA[t] / fmaxf(pB[t], 1e-9f);
  __syncthreads();
  if (t == 0) {
    float a = 0.0f;
    for (int i = 0; i < BT_; ++i) a += s[i];
    out[0] = a;
  }
}

// ---------------------------------------------------------------------------
extern "C" void kernel_launch(void* const* d_in, const int* in_sizes, int n_in,
                              void* d_out, int out_size, void* d_ws, size_t ws_size,
                              hipStream_t stream) {
  (void)in_sizes; (void)n_in; (void)out_size; (void)ws_size;

  const float* x      = (const float*)d_in[0];
  const float* W      = (const float*)d_in[1];
  const float* bias   = (const float*)d_in[2];
  const float* cw     = (const float*)d_in[3];
  const int*   labels = (const int*)d_in[4];
  const int*   mask   = (const int*)d_in[5];

  const size_t qk_elems = (size_t)BT_ * C_ * L_ * DQK; // 6,291,456 bf16 each
  char* ws = (char*)d_ws;
  __bf16* Qb = (__bf16*)ws;
  __bf16* Kb = (__bf16*)(ws + qk_elems * sizeof(__bf16));
  float*  pA = (float*)(ws + 2 * qk_elems * sizeof(__bf16));
  float*  pB = pA + BT_;

  hipMemsetAsync(pA, 0, 2 * BT_ * sizeof(float), stream);

  proj_qk_kernel<<<(M1 / 128) * (N1 / 64), 256, 0, stream>>>(x, W, bias, Qb, Kb);
  logits_loss_kernel<<<BT_ * (L_ / 64) * (L_ / 64), 256, 0, stream>>>(
      Qb, Kb, labels, mask, cw, pA, pB);
  final_reduce_kernel<<<1, 32, 0, stream>>>(pA, pB, (float*)d_out);
}
